// DHTR_70514773066009
// MI455X (gfx1250) — matrix operand order repaired and checked by reading, not compiled, not run on hardware
//
#include <hip/hip_runtime.h>
#include <hip/hip_bf16.h>
#include <math.h>

#define B_    64
#define SRC_  65
#define L_    128
#define T_    127
#define E_    256
#define H_    128
#define A_    256
#define COLS_ 80
#define CELLS_ 12800

typedef __bf16 bf16_t;
typedef __attribute__((ext_vector_type(16))) __bf16 v16bf;
typedef __attribute__((ext_vector_type(8)))  __bf16 v8bf;
typedef __attribute__((ext_vector_type(8)))  float  v8f;
typedef unsigned int v4u_t __attribute__((ext_vector_type(4)));
typedef int          v4i_t __attribute__((ext_vector_type(4)));
typedef int          v8i_t __attribute__((ext_vector_type(8)));

__device__ __forceinline__ float sigf(float x) { return 1.0f / (1.0f + __expf(-x)); }

// ---------------- generic f32 -> bf16 convert ----------------
__global__ void k_f2bf(const float* __restrict__ s, bf16_t* __restrict__ d, int n) {
  int i = blockIdx.x * blockDim.x + threadIdx.x;
  if (i < n) d[i] = (bf16_t)s[i];
}

__global__ void k_bias_sum(const float* __restrict__ a, const float* __restrict__ b,
                           float* __restrict__ o, int n) {
  int i = blockIdx.x * blockDim.x + threadIdx.x;
  if (i < n) o[i] = a[i] + b[i];
}

// ---------------- gathers ----------------
__global__ __launch_bounds__(256) void k_gather_x(const int* __restrict__ src_ids,
    const float* __restrict__ loc_emb, bf16_t* __restrict__ xbf) {
  int row = blockIdx.x, e = threadIdx.x;
  int cid = src_ids[row];
  xbf[(long)row * E_ + e] = (bf16_t)loc_emb[(long)cid * E_ + e];
}

__global__ __launch_bounds__(256) void k_gather_sp(const int* __restrict__ src_ids,
    const float* __restrict__ sp_emb, bf16_t* __restrict__ spbf) {
  int row = blockIdx.x, e = threadIdx.x;
  int cid = src_ids[row];
  int r2 = cid / COLS_, c2 = cid % COLS_;
  int d = (int)floorf(sqrtf((float)(r2 * r2 + c2 * c2)));
  if (d > 255) d = 255;
  spbf[(long)row * E_ + e] = (bf16_t)sp_emb[(long)d * E_ + e];
}

__global__ __launch_bounds__(256) void k_gather_tp(const int* __restrict__ obs_steps,
    const float* __restrict__ tp_emb, bf16_t* __restrict__ tpbf) {
  int row = blockIdx.x, e = threadIdx.x;
  int t = row / SRC_, s = row % SRC_;
  int dist = (t + 1) - obs_steps[s];
  if (dist < 0) dist = -dist;
  if (dist > 255) dist = 255;
  tpbf[(long)row * E_ + e] = (bf16_t)tp_emb[(long)dist * E_ + e];
}

__global__ __launch_bounds__(256) void k_gather_rcat(const int* __restrict__ src_ids,
    const int* __restrict__ pre_ids, const int* __restrict__ suc_ids,
    const float* __restrict__ loc_emb, bf16_t* __restrict__ rcat) {
  int row = blockIdx.x, e = threadIdx.x;
  int b = row / L_, l = row % L_;
  int cp = src_ids[b * SRC_ + pre_ids[l]];
  int cs = src_ids[b * SRC_ + suc_ids[l]];
  rcat[(long)row * (2 * E_) + e]      = (bf16_t)loc_emb[(long)cp * E_ + e];
  rcat[(long)row * (2 * E_) + E_ + e] = (bf16_t)loc_emb[(long)cs * E_ + e];
}

__global__ __launch_bounds__(256) void k_gather_prev(const int* __restrict__ src_ids,
    const int* __restrict__ trg_ids, const float* __restrict__ loc_emb,
    float* __restrict__ prevA) {
  int row = blockIdx.x, e = threadIdx.x;
  int b = row / L_, l = row % L_;
  int cid = (l == 1) ? src_ids[b * SRC_] : trg_ids[b * L_ + l];
  prevA[(long)row * E_ + e] = loc_emb[(long)cid * E_ + e];
}

// ---------------- WMMA bf16 GEMM: C[M,N] = A[M,K] @ W[N,K]^T (+bias) ----------------
// The 64 x K A-panel is DMA'd once into LDS by the Tensor Data Mover (wave 0 issues
// the descriptor; OOB rows beyond M are hardware zero-filled), then the K-loop runs
// barrier-free: 2 global B-fragment loads + 8 ds loads + 4 WMMA per 32-wide K step.
// flags bit0 = accumulate into C, bit1 = add bias[n]. Optional per-row output offsets.
__global__ __launch_bounds__(256) void k_gemm_bf16(
    const bf16_t* __restrict__ A, int lda,
    const bf16_t* __restrict__ W, int ldw,
    const float* __restrict__ bias,
    float* __restrict__ C, long ldc,
    const long* __restrict__ row_off,
    int M, int N, int K, int flags)
{
  extern __shared__ bf16_t Ap[];   // [64][K] bf16, K <= 512 -> <= 64 KB
  const int tid  = threadIdx.x;
  const int wave = tid >> 5;
  const int lane = tid & 31;
  const int m0 = blockIdx.y * 64;
  const int n0 = (blockIdx.x * 8 + wave) * 16;

#if __has_builtin(__builtin_amdgcn_tensor_load_to_lds)
  if (wave == 0) {
    // Build Tensor DMA Descriptor (D#): 2D tile, 2-byte elements.
    unsigned long long ga = (unsigned long long)(const void*)(A + (long)m0 * lda);
    unsigned ldsa = (unsigned)(unsigned long long)(void*)Ap; // low bits = LDS byte addr
    unsigned dim1 = (unsigned)(M - m0);                      // remaining rows (OOB -> 0)
    v4u_t g0; v8i_t g1; v4i_t gz4; v8i_t gz8;
#pragma unroll
    for (int i = 0; i < 4; ++i) gz4[i] = 0;
#pragma unroll
    for (int i = 0; i < 8; ++i) gz8[i] = 0;
    g0[0] = 1u;                                            // count=1, user mode
    g0[1] = ldsa;                                          // lds_addr
    g0[2] = (unsigned)(ga & 0xffffffffull);                // global_addr[31:0]
    g0[3] = (unsigned)((ga >> 32) & 0x1ffffffull) | (2u << 30); // addr[56:32] | type=2
    g1[0] = (int)(1u << 16);                               // data_size = 1 (2 bytes)
    g1[1] = (int)(((unsigned)K & 0xffffu) << 16);          // tensor_dim0[15:0]
    g1[2] = (int)((dim1 & 0xffffu) << 16);                 // tensor_dim0[31:16]=0 | tensor_dim1[15:0]
    g1[3] = (int)(((unsigned)K & 0xffffu) << 16);          // tensor_dim1[31:16]=0 | tile_dim0=K
    g1[4] = 64;                                            // tile_dim1=64, tile_dim2=0
    g1[5] = (int)(unsigned)lda;                            // tensor_dim0_stride[31:0]
    g1[6] = 0;                                             // stride0[47:32] | stride1[15:0]
    g1[7] = 0;                                             // stride1[47:16]
#if defined(__clang_major__) && __clang_major__ >= 23
    __builtin_amdgcn_tensor_load_to_lds(g0, g1, gz4, gz4, gz8, 0);
#else
    __builtin_amdgcn_tensor_load_to_lds(g0, g1, gz4, gz4, 0);
#endif
    __builtin_amdgcn_s_wait_tensorcnt(0);
  }
#else
  // Fallback: cooperative vector staging of the whole panel.
  for (int idx = tid; idx < 64 * (K >> 3); idx += 256) {
    int r = idx / (K >> 3), sg = (idx % (K >> 3)) << 3;
    int gm = m0 + r;
    v8bf val;
    if (gm < M) {
      val = *(const v8bf*)(A + (long)gm * lda + sg);
    } else {
#pragma unroll
      for (int q = 0; q < 8; ++q) val[q] = (bf16_t)0.0f;
    }
    *(v8bf*)(Ap + r * K + sg) = val;
  }
#endif
  __syncthreads();
  if (n0 >= N) return;

  v8f acc[4];
#pragma unroll
  for (int mt = 0; mt < 4; ++mt)
#pragma unroll
    for (int v = 0; v < 8; ++v) acc[mt][v] = 0.0f;

  const int koff = (lane >> 4) << 3;               // A frag: k-run base 0/8 (+16)
  const int kb   = (lane >> 4) << 3;               // B frag: k-run base 0/8 (+16)
  const bf16_t* wrow = W + (long)(n0 + (lane & 15)) * ldw;
  const bf16_t* arow = Ap + (lane & 15) * K;

  for (int k0 = 0; k0 < K; k0 += 32) {
    const bf16_t* bp = wrow + k0 + kb;
    v16bf bfrag;
    ((v8bf*)&bfrag)[0] = *(const v8bf*)(bp);
    ((v8bf*)&bfrag)[1] = *(const v8bf*)(bp + 16);
    if (k0 + 32 < K) __builtin_prefetch((const void*)(bp + 32), 0, 1);
#pragma unroll
    for (int mt = 0; mt < 4; ++mt) {
      const bf16_t* ap = arow + mt * 16 * K + k0 + koff;
      v16bf afrag;
      ((v8bf*)&afrag)[0] = *(const v8bf*)(ap);
      ((v8bf*)&afrag)[1] = *(const v8bf*)(ap + 16);
      acc[mt] = __builtin_amdgcn_wmma_f32_16x16x32_bf16(
          false, afrag, false, bfrag, (short)0, acc[mt], false, false);
    }
  }

  const int nl = lane & 15;
  const int mh = (lane >> 4) << 3;
  const int gn = n0 + nl;
  const float bv = ((flags & 2) && bias) ? bias[gn] : 0.0f;
#pragma unroll
  for (int mt = 0; mt < 4; ++mt) {
#pragma unroll
    for (int v = 0; v < 8; ++v) {
      int gm = m0 + mt * 16 + mh + v;
      if (gm < M && gn < N) {
        float val = acc[mt][v] + bv;
        long o = row_off ? (row_off[gm] + gn) : ((long)gm * ldc + gn);
        if (flags & 1) C[o] += val; else C[o] = val;
      }
    }
  }
}

// ---------------- BiLSTM encoder: one WG per batch element, both directions ----------------
__global__ __launch_bounds__(256) void k_encoder(
    const float* __restrict__ preF, const float* __restrict__ preB,
    const float* __restrict__ WhhF, const float* __restrict__ WhhB,
    float* __restrict__ enc_s, bf16_t* __restrict__ enc_sbf)
{
  __shared__ float hl[2][H_];
  const int b = blockIdx.x;
  const int tid = threadIdx.x;
  const int dir = tid >> 7;
  const int j = tid & (H_ - 1);
  const float* pre = dir ? preB : preF;
  const float* Whh = dir ? WhhB : WhhF;
  float c = 0.0f;
  hl[dir][j] = 0.0f;
  __syncthreads();
  for (int k = 0; k < SRC_; ++k) {
    const int s = dir ? (SRC_ - 1 - k) : k;
    const float* prow = pre + ((long)b * SRC_ + s) * (4 * H_);
    float zi = prow[j], zf = prow[H_ + j], zg = prow[2 * H_ + j], zo = prow[3 * H_ + j];
    const float* wi = Whh + (long)j * H_;
    const float* wf = Whh + (long)(H_ + j) * H_;
    const float* wg = Whh + (long)(2 * H_ + j) * H_;
    const float* wo = Whh + (long)(3 * H_ + j) * H_;
    for (int kk = 0; kk < H_; ++kk) {
      float hv = hl[dir][kk];
      zi = fmaf(hv, wi[kk], zi);
      zf = fmaf(hv, wf[kk], zf);
      zg = fmaf(hv, wg[kk], zg);
      zo = fmaf(hv, wo[kk], zo);
    }
    c = sigf(zf) * c + sigf(zi) * tanhf(zg);
    float h = sigf(zo) * tanhf(c);
    __syncthreads();
    hl[dir][j] = h;
    long eo = ((long)b * SRC_ + s) * (2 * H_) + dir * H_ + j;
    enc_s[eo] = h;
    enc_sbf[eo] = (bf16_t)h;
    __syncthreads();
  }
}

// ---------------- decoder core: one WG per batch element, all 127 steps ----------------
__global__ __launch_bounds__(256) void k_dec_core(
    const float* __restrict__ enc_s, const float* __restrict__ swq,
    const float* __restrict__ wp_all, const float* __restrict__ prevA,
    const float* __restrict__ r_all, const float* __restrict__ WH,
    const float* __restrict__ v_w, const float* __restrict__ cat_W,
    const float* __restrict__ cat_b, const float* __restrict__ dWih,
    const float* __restrict__ dWhh, const float* __restrict__ dbih,
    const float* __restrict__ dbhh, bf16_t* __restrict__ hsig)
{
  __shared__ float hrow[2 * H_];
  __shared__ float qv[A_];
  __shared__ float red[264];
  __shared__ float att[SRC_ + 3];
  __shared__ float catv[3 * E_];
  __shared__ float xin[E_];
  __shared__ float zrow[4 * E_];
  const int b = blockIdx.x;
  const int tid = threadIdx.x;
  hrow[tid] = enc_s[((long)b * SRC_ + (SRC_ - 1)) * (2 * H_) + tid];
  float c = 0.0f;
  __syncthreads();
  for (int t = 1; t < L_; ++t) {
    { // q = h[b] @ WH^T
      float a = 0.0f;
      const float* w = WH + (long)tid * E_;
      for (int kk = 0; kk < E_; ++kk) a = fmaf(hrow[kk], w[kk], a);
      qv[tid] = a;
    }
    __syncthreads();
    // attention scores: 65 positions x 4 partial lanes
    for (int it = tid; it < SRC_ * 4; it += 256) {
      const int s = it >> 2, part = (it & 3) << 6;
      const float* sw = swq + ((long)b * SRC_ + s) * A_ + part;
      const float* wp = wp_all + ((long)(t - 1) * SRC_ + s) * A_ + part;
      const float* qq = qv + part;
      const float* vv = v_w + part;
      float p = 0.0f;
      for (int a = 0; a < 64; ++a) p = fmaf(vv[a], tanhf(qq[a] + sw[a] + wp[a]), p);
      red[it] = p;
    }
    __syncthreads();
    if (tid < SRC_) att[tid] = red[tid * 4] + red[tid * 4 + 1] + red[tid * 4 + 2] + red[tid * 4 + 3];
    __syncthreads();
    if (tid == 0) { // softmax over 65
      float mx = att[0];
      for (int s = 1; s < SRC_; ++s) mx = fmaxf(mx, att[s]);
      float sm = 0.0f;
      for (int s = 0; s < SRC_; ++s) { float e = __expf(att[s] - mx); att[s] = e; sm += e; }
      float inv = 1.0f / sm;
      for (int s = 0; s < SRC_; ++s) att[s] *= inv;
    }
    __syncthreads();
    { // context + build cat vector [prev, r, e]
      float e = 0.0f;
      const float* es = enc_s + (long)b * SRC_ * (2 * H_) + tid;
      for (int s = 0; s < SRC_; ++s) e = fmaf(att[s], es[(long)s * (2 * H_)], e);
      catv[2 * E_ + tid] = e;
      catv[tid]      = prevA[((long)b * L_ + t) * E_ + tid];
      catv[E_ + tid] = r_all[((long)b * L_ + t) * E_ + tid];
    }
    __syncthreads();
    { // xin = cat @ cat_W^T + cat_b
      float a = cat_b[tid];
      const float* w = cat_W + (long)tid * (3 * E_);
      for (int kk = 0; kk < 3 * E_; ++kk) a = fmaf(w[kk], catv[kk], a);
      xin[tid] = a;
    }
    __syncthreads();
    // z = xin @ dWih^T + h @ dWhh^T + biases
    for (int r = 0; r < 4; ++r) {
      const int n = (r << 8) + tid;
      float a = dbih[n] + dbhh[n];
      const float* wi = dWih + (long)n * E_;
      const float* wh = dWhh + (long)n * E_;
      for (int kk = 0; kk < E_; ++kk) {
        a = fmaf(wi[kk], xin[kk], a);
        a = fmaf(wh[kk], hrow[kk], a);
      }
      zrow[n] = a;
    }
    __syncthreads();
    {
      const float ig = zrow[tid], fg = zrow[E_ + tid], gg = zrow[2 * E_ + tid], og = zrow[3 * E_ + tid];
      c = sigf(fg) * c + sigf(ig) * tanhf(gg);
      const float h = sigf(og) * tanhf(c);
      __syncthreads();
      hrow[tid] = h;
      hsig[((long)(t - 1) * B_ + b) * E_ + tid] = (bf16_t)sigf(h);
    }
    __syncthreads();
  }
}

// ---------------- classifier support ----------------
__global__ void k_row_offsets(long* __restrict__ ro) {
  int m = blockIdx.x * blockDim.x + threadIdx.x;
  if (m < T_ * B_) {
    int t = (m >> 6) + 1, b = m & 63;
    ro[m] = ((long)b * L_ + t) * CELLS_;
  }
}

__global__ __launch_bounds__(256) void k_row_stats(
    const float* __restrict__ out_prob, float* __restrict__ rowMax,
    float* __restrict__ rowSum, float* __restrict__ out_ids,
    const int* __restrict__ trg_ids)
{
  const int m = blockIdx.x;
  const int tid = threadIdx.x;
  const int t = (m >> 6) + 1;
  const int b = m & 63;
  const bool masked = (t & 1) && (t != L_ - 1);
  if (!masked) {
    if (tid == 0) out_ids[(long)b * L_ + t] = (float)trg_ids[b * L_ + t];
    return;
  }
  __shared__ float s0[256];
  __shared__ float s1[256];
  __shared__ int   si[256];
  const float* row = out_prob + ((long)b * L_ + t) * CELLS_;
  float mx = -3.402823466e38f;
  for (int ci = tid; ci < CELLS_; ci += 256) mx = fmaxf(mx, row[ci]);
  s0[tid] = mx;
  __syncthreads();
  for (int w = 128; w > 0; w >>= 1) {
    if (tid < w) s0[tid] = fmaxf(s0[tid], s0[tid + w]);
    __syncthreads();
  }
  mx = s0[0];
  __syncthreads();
  float sm = 0.0f, bv = -3.402823466e38f; int bi = 0;
  for (int ci = tid; ci < CELLS_; ci += 256) {
    float x = row[ci];
    sm += __expf(x - mx);
    if (x > bv) { bv = x; bi = ci; }
  }
  s0[tid] = bv; s1[tid] = sm; si[tid] = bi;
  __syncthreads();
  for (int w = 128; w > 0; w >>= 1) {
    if (tid < w) {
      s1[tid] += s1[tid + w];
      if (s0[tid + w] > s0[tid] || (s0[tid + w] == s0[tid] && si[tid + w] < si[tid])) {
        s0[tid] = s0[tid + w]; si[tid] = si[tid + w];
      }
    }
    __syncthreads();
  }
  if (tid == 0) {
    rowMax[m] = mx;
    rowSum[m] = s1[0];
    out_ids[(long)b * L_ + t] = (float)si[0];
  }
}

__global__ __launch_bounds__(256) void k_finalize(
    float* __restrict__ out_prob, const float* __restrict__ rowMax,
    const float* __restrict__ rowSum)
{
  const int m = blockIdx.y;
  const int t = (m >> 6) + 1;
  const int b = m & 63;
  const bool masked = (t & 1) && (t != L_ - 1);
  const long o = ((long)b * L_ + t) * CELLS_ + (long)blockIdx.x * 256 + threadIdx.x;
  float v = 0.0f;
  if (masked) v = __expf(out_prob[o] - rowMax[m]) / rowSum[m];
  out_prob[o] = v;
}

__global__ __launch_bounds__(256) void k_init_out(float* __restrict__ out_prob) {
  const int b = blockIdx.y;
  const long cell = (long)blockIdx.x * 256 + threadIdx.x;
  out_prob[((long)b * L_) * CELLS_ + cell] = 0.0f; // step 0 slab
}

__global__ void k_init_ids(const int* __restrict__ src_ids, float* __restrict__ out_ids) {
  int b = threadIdx.x;
  if (b < B_) out_ids[(long)b * L_] = (float)src_ids[b * SRC_];
}

// ---------------- host ----------------
extern "C" void kernel_launch(void* const* d_in, const int* in_sizes, int n_in,
                              void* d_out, int out_size, void* d_ws, size_t ws_size,
                              hipStream_t stream) {
  (void)in_sizes; (void)n_in; (void)out_size; (void)ws_size;
  const int*   src_ids   = (const int*)d_in[0];
  const int*   trg_ids   = (const int*)d_in[1];
  const int*   obs_steps = (const int*)d_in[2];
  const int*   pre_ids   = (const int*)d_in[3];
  const int*   suc_ids   = (const int*)d_in[4];
  const float* loc_emb   = (const float*)d_in[6];
  const float* sp_emb    = (const float*)d_in[7];
  const float* tp_emb    = (const float*)d_in[8];
  const float* eWihF = (const float*)d_in[9];
  const float* eWhhF = (const float*)d_in[10];
  const float* ebihF = (const float*)d_in[11];
  const float* ebhhF = (const float*)d_in[12];
  const float* eWihB = (const float*)d_in[13];
  const float* eWhhB = (const float*)d_in[14];
  const float* ebihB = (const float*)d_in[15];
  const float* ebhhB = (const float*)d_in[16];
  const float* WH    = (const float*)d_in[17];
  const float* WS    = (const float*)d_in[18];
  const float* WP    = (const float*)d_in[19];
  const float* WQ    = (const float*)d_in[20];
  const float* v_w   = (const float*)d_in[21];
  const float* cons_W = (const float*)d_in[22];
  const float* cons_b = (const float*)d_in[23];
  const float* cat_W  = (const float*)d_in[24];
  const float* cat_b  = (const float*)d_in[25];
  const float* dWih   = (const float*)d_in[26];
  const float* dWhh   = (const float*)d_in[27];
  const float* dbih   = (const float*)d_in[28];
  const float* dbhh   = (const float*)d_in[29];
  const float* cls_W  = (const float*)d_in[30];
  const float* cls_b  = (const float*)d_in[31];

  float* out_prob = (float*)d_out;
  float* out_ids  = out_prob + (size_t)B_ * L_ * CELLS_;

  char* wsb = (char*)d_ws;
  size_t off = 0;
  auto alloc = [&](size_t bytes) -> void* {
    void* p = wsb + off;
    off = (off + bytes + 255) & ~(size_t)255;
    return p;
  };
  bf16_t* xbf    = (bf16_t*)alloc((size_t)B_ * SRC_ * E_ * 2);
  bf16_t* spbf   = (bf16_t*)alloc((size_t)B_ * SRC_ * E_ * 2);
  bf16_t* tpbf   = (bf16_t*)alloc((size_t)T_ * SRC_ * E_ * 2);
  bf16_t* rcat   = (bf16_t*)alloc((size_t)B_ * L_ * 2 * E_ * 2);
  float*  prevA  = (float*) alloc((size_t)B_ * L_ * E_ * 4);
  float*  r_all  = (float*) alloc((size_t)B_ * L_ * E_ * 4);
  float*  preF   = (float*) alloc((size_t)B_ * SRC_ * 4 * H_ * 4);
  float*  preB   = (float*) alloc((size_t)B_ * SRC_ * 4 * H_ * 4);
  float*  enc_s  = (float*) alloc((size_t)B_ * SRC_ * 2 * H_ * 4);
  bf16_t* enc_sbf= (bf16_t*)alloc((size_t)B_ * SRC_ * 2 * H_ * 2);
  float*  swq    = (float*) alloc((size_t)B_ * SRC_ * A_ * 4);
  float*  wp_all = (float*) alloc((size_t)T_ * SRC_ * A_ * 4);
  bf16_t* hsig   = (bf16_t*)alloc((size_t)T_ * B_ * E_ * 2);
  bf16_t* WihFb  = (bf16_t*)alloc((size_t)4 * H_ * E_ * 2);
  bf16_t* WihBb  = (bf16_t*)alloc((size_t)4 * H_ * E_ * 2);
  bf16_t* WSb    = (bf16_t*)alloc((size_t)A_ * E_ * 2);
  bf16_t* WQb    = (bf16_t*)alloc((size_t)A_ * E_ * 2);
  bf16_t* WPb    = (bf16_t*)alloc((size_t)A_ * E_ * 2);
  bf16_t* consWb = (bf16_t*)alloc((size_t)E_ * 2 * E_ * 2);
  bf16_t* clsWb  = (bf16_t*)alloc((size_t)CELLS_ * E_ * 2);
  float*  bsumF  = (float*) alloc((size_t)4 * H_ * 4);
  float*  bsumB  = (float*) alloc((size_t)4 * H_ * 4);
  long*   rowOff = (long*)  alloc((size_t)T_ * B_ * 8);
  float*  rowMax = (float*) alloc((size_t)T_ * B_ * 4);
  float*  rowSum = (float*) alloc((size_t)T_ * B_ * 4);

  auto cvt = [&](const float* s, bf16_t* d, int n) {
    k_f2bf<<<dim3((n + 255) / 256), dim3(256), 0, stream>>>(s, d, n);
  };
  auto gemm = [&](const bf16_t* Ap, int lda, const bf16_t* Wp, int ldw,
                  const float* bias, float* Cp, long ldc, const long* ro,
                  int M, int N, int K, int flags) {
    size_t shbytes = (size_t)64 * K * sizeof(bf16_t);
    k_gemm_bf16<<<dim3((N + 127) / 128, (M + 63) / 64), dim3(256), shbytes, stream>>>(
        Ap, lda, Wp, ldw, bias, Cp, ldc, ro, M, N, K, flags);
  };

  // output init (step-0 slab + ids col 0)
  k_init_out<<<dim3(CELLS_ / 256, B_), dim3(256), 0, stream>>>(out_prob);
  k_init_ids<<<dim3(1), dim3(B_), 0, stream>>>(src_ids, out_ids);

  // weight converts
  cvt(eWihF, WihFb, 4 * H_ * E_);
  cvt(eWihB, WihBb, 4 * H_ * E_);
  cvt(WS, WSb, A_ * E_);
  cvt(WQ, WQb, A_ * E_);
  cvt(WP, WPb, A_ * E_);
  cvt(cons_W, consWb, E_ * 2 * E_);
  cvt(cls_W, clsWb, CELLS_ * E_);
  k_bias_sum<<<dim3(2), dim3(256), 0, stream>>>(ebihF, ebhhF, bsumF, 4 * H_);
  k_bias_sum<<<dim3(2), dim3(256), 0, stream>>>(ebihB, ebhhB, bsumB, 4 * H_);

  // gathers
  k_gather_x   <<<dim3(B_ * SRC_), dim3(256), 0, stream>>>(src_ids, loc_emb, xbf);
  k_gather_sp  <<<dim3(B_ * SRC_), dim3(256), 0, stream>>>(src_ids, sp_emb, spbf);
  k_gather_tp  <<<dim3(T_ * SRC_), dim3(256), 0, stream>>>(obs_steps, tp_emb, tpbf);
  k_gather_rcat<<<dim3(B_ * L_),  dim3(256), 0, stream>>>(src_ids, pre_ids, suc_ids, loc_emb, rcat);
  k_gather_prev<<<dim3(B_ * L_),  dim3(256), 0, stream>>>(src_ids, trg_ids, loc_emb, prevA);

  // WMMA GEMMs: encoder input precompute, wp, r
  gemm(xbf, E_, WihFb, E_, bsumF, preF, 4 * H_, nullptr, B_ * SRC_, 4 * H_, E_, 2);
  gemm(xbf, E_, WihBb, E_, bsumB, preB, 4 * H_, nullptr, B_ * SRC_, 4 * H_, E_, 2);
  gemm(tpbf, E_, WPb, E_, nullptr, wp_all, A_, nullptr, T_ * SRC_, A_, E_, 0);
  gemm(rcat, 2 * E_, consWb, 2 * E_, cons_b, r_all, E_, nullptr, B_ * L_, E_, 2 * E_, 2);

  // encoder recurrence (64 independent WGs)
  k_encoder<<<dim3(B_), dim3(256), 0, stream>>>(preF, preB, eWhhF, eWhhB, enc_s, enc_sbf);

  // swq = enc_s @ WS^T + sp @ WQ^T
  gemm(enc_sbf, 2 * H_, WSb, E_, nullptr, swq, A_, nullptr, B_ * SRC_, A_, E_, 0);
  gemm(spbf,    E_,     WQb, E_, nullptr, swq, A_, nullptr, B_ * SRC_, A_, E_, 1);

  // decoder recurrence (64 independent WGs, no grid sync needed)
  k_dec_core<<<dim3(B_), dim3(256), 0, stream>>>(enc_s, swq, wp_all, prevA, r_all,
      WH, v_w, cat_W, cat_b, dWih, dWhh, dbih, dbhh, hsig);

  // classifier: big WMMA GEMM straight into d_out (row-remapped), then softmax in place
  k_row_offsets<<<dim3((T_ * B_ + 255) / 256), dim3(256), 0, stream>>>(rowOff);
  gemm(hsig, E_, clsWb, E_, cls_b, out_prob, 0L, rowOff, T_ * B_, CELLS_, E_, 2);
  k_row_stats<<<dim3(T_ * B_), dim3(256), 0, stream>>>(out_prob, rowMax, rowSum, out_ids, trg_ids);
  k_finalize <<<dim3(CELLS_ / 256, T_ * B_), dim3(256), 0, stream>>>(out_prob, rowMax, rowSum);
}
